// LSTM_20298015441555
// MI455X (gfx1250) — matrix operand order repaired and checked by reading, not compiled, not run on hardware
//
#include <hip/hip_runtime.h>

typedef __bf16 bf16_t;
typedef __attribute__((ext_vector_type(16))) __bf16 v16bf;
typedef __attribute__((ext_vector_type(8)))  __bf16 v8bf;
typedef __attribute__((ext_vector_type(8)))  float  v8f;

#define T_STEPS 512
#define BATCH   32
#define INP     1024
#define HID     1024
#define NLAYER  2
#define GATES   (4 * HID)
#define NWG     (2 * (HID / 16))   // 128 workgroups: 64 per layer

// ---------------------------------------------------------------------------
// WMMA fragment loaders (V_WMMA_F32_16X16X32_BF16, wave32)
//
// A (16x32, MxK) bf16, row-major memory, stride ld:
//   lane L: row M = L&15
//     elems 0..7  = K[(L>>4)*8      .. +7]   (VGPR0..3)
//     elems 8..15 = K[16 + (L>>4)*8 .. +7]   (VGPR4..7)
// B (32x16, KxN) bf16 where B[k][n] = W[n][k], W row-major:
//   lane L: col N = L&15, elems 0..7 = K[(L>>4)*8 .. +7], elems 8..15 = K[16 + (L>>4)*8 .. +7]
// ---------------------------------------------------------------------------

__device__ __forceinline__ v16bf load_A_frag(const bf16_t* __restrict__ base,
                                             int row0, int kb, int ld, int lane) {
    const int r  = lane & 15;
    const int hi = lane >> 4;
    const bf16_t* p = base + (size_t)(row0 + r) * (size_t)ld + kb + hi * 8;
    v8bf lo = *(const v8bf*)(p);
    v8bf hv = *(const v8bf*)(p + 16);
    v16bf a;
#pragma unroll
    for (int i = 0; i < 8; ++i) { a[i] = lo[i]; a[i + 8] = hv[i]; }
    return a;
}

__device__ __forceinline__ v16bf load_B_frag(const bf16_t* __restrict__ wrows,
                                             int kb, int lane) {
    const int n  = lane & 15;
    const int hi = lane >> 4;
    const bf16_t* p = wrows + (size_t)n * (size_t)HID + kb + hi * 8;
    v8bf b0 = *(const v8bf*)(p);
    v8bf b1 = *(const v8bf*)(p + 16);
    v16bf b;
#pragma unroll
    for (int i = 0; i < 8; ++i) { b[i] = b0[i]; b[i + 8] = b1[i]; }
    return b;
}

__device__ __forceinline__ float sigmoid_f(float x) {
    return 1.0f / (1.0f + __expf(-x));
}

// ---------------------------------------------------------------------------
// Device-wide generation barrier. All NWG blocks arrive; last arriver resets
// the counter and release-publishes generation = it+1. h/c/out stores reach
// the device-coherent L2 before publication (threadfence + ACQ_REL arrive);
// the acquire spin invalidates WGP-local caches before the next phase reads.
// ---------------------------------------------------------------------------
__device__ __forceinline__ void grid_sync(unsigned* __restrict__ count,
                                          unsigned* __restrict__ gen,
                                          unsigned it) {
    __threadfence();      // each thread: publish its global stores device-wide
    __syncthreads();      // all block stores ordered before the arrive below
    if (threadIdx.x == 0) {
        unsigned prev = __hip_atomic_fetch_add(count, 1u, __ATOMIC_ACQ_REL,
                                               __HIP_MEMORY_SCOPE_AGENT);
        if (prev == (unsigned)(NWG - 1)) {
            __hip_atomic_store(count, 0u, __ATOMIC_RELAXED,
                               __HIP_MEMORY_SCOPE_AGENT);
            __hip_atomic_store(gen, it + 1u, __ATOMIC_RELEASE,
                               __HIP_MEMORY_SCOPE_AGENT);
        } else {
            while (__hip_atomic_load(gen, __ATOMIC_ACQUIRE,
                                     __HIP_MEMORY_SCOPE_AGENT) < it + 1u)
                __builtin_amdgcn_s_sleep(2);
        }
    }
    __syncthreads();      // release the rest of the block past the barrier
}

// ---------------------------------------------------------------------------
// One (layer, timestep) for this WG's 16 hidden columns: 8 waves, each wave
// one (gate, m-half) 16x16 tile, K = 1024(x) + 1024(h), fused cell update.
// ---------------------------------------------------------------------------
__device__ __forceinline__ void wg_cell(const bf16_t* __restrict__ xin,
                                        const bf16_t* __restrict__ hprev,
                                        const bf16_t* __restrict__ Wih_l,
                                        const bf16_t* __restrict__ Whh_l,
                                        const float*  __restrict__ bih_l,
                                        const float*  __restrict__ bhh_l,
                                        float*        __restrict__ cst_l,
                                        bf16_t*       __restrict__ hbout,
                                        float*        __restrict__ memout,
                                        float*        __restrict__ outT_h,
                                        float*        __restrict__ outT_c,
                                        float*        __restrict__ gtile,
                                        int n0) {
    const int tid  = threadIdx.x;
    const int lane = tid & 31;
    const int wave = tid >> 5;
    const int gate = wave >> 1;   // 0..3 (i,f,g,o)
    const int m    = wave & 1;    // batch half: rows m*16 .. m*16+15

    const bf16_t* wih_rows = Wih_l + (size_t)(gate * HID + n0) * (size_t)INP;
    const bf16_t* whh_rows = Whh_l + (size_t)(gate * HID + n0) * (size_t)HID;

    v8f acc = {};
#pragma unroll 4
    for (int kb = 0; kb < INP; kb += 32) {
        v16bf a = load_A_frag(xin, m * 16, kb, INP, lane);
        v16bf b = load_B_frag(wih_rows, kb, lane);
        acc = __builtin_amdgcn_wmma_f32_16x16x32_bf16(false, a, false, b,
                                                      (short)0, acc, false, false);
    }
#pragma unroll 4
    for (int kb = 0; kb < HID; kb += 32) {
        v16bf a = load_A_frag(hprev, m * 16, kb, HID, lane);
        v16bf b = load_B_frag(whh_rows, kb, lane);
        acc = __builtin_amdgcn_wmma_f32_16x16x32_bf16(false, a, false, b,
                                                      (short)0, acc, false, false);
    }

    // Spill D tile: VGPR r -> (M = (lane>>4)*8 + r, N = lane&15)
    {
        float* tp = gtile + wave * 256;
        const int col   = lane & 15;
        const int rbase = (lane >> 4) * 8;
#pragma unroll
        for (int r = 0; r < 8; ++r) tp[(rbase + r) * 16 + col] = acc[r];
    }
    __syncthreads();

    // Cell update: 32 batch rows x 16 cols = 512 elems, 2 per thread.
    for (int e = tid; e < 512; e += 256) {
        const int b   = e >> 4;
        const int col = e & 15;
        const int mh  = b >> 4;
        const int rr  = b & 15;
        const int gc  = n0 + col;

        const float gi = gtile[(0 * 2 + mh) * 256 + rr * 16 + col] + bih_l[0 * HID + gc] + bhh_l[0 * HID + gc];
        const float gf = gtile[(1 * 2 + mh) * 256 + rr * 16 + col] + bih_l[1 * HID + gc] + bhh_l[1 * HID + gc];
        const float gg = gtile[(2 * 2 + mh) * 256 + rr * 16 + col] + bih_l[2 * HID + gc] + bhh_l[2 * HID + gc];
        const float go = gtile[(3 * 2 + mh) * 256 + rr * 16 + col] + bih_l[3 * HID + gc] + bhh_l[3 * HID + gc];

        const float iv = sigmoid_f(gi);
        const float fv = sigmoid_f(gf);
        const float gv = tanhf(gg);
        const float ov = sigmoid_f(go);

        const size_t idx = (size_t)b * HID + gc;
        const float cn = fv * cst_l[idx] + iv * gv;
        const float hn = ov * tanhf(cn);
        cst_l[idx] = cn;
        hbout[idx] = (bf16_t)hn;
        if (memout) memout[idx] = hn;
        if (outT_h) { outT_h[idx] = hn; outT_c[idx] = cn; }
    }
}

// ---------------------------------------------------------------------------
// Persistent kernel: 128 WGs, software-pipelined layers.
// Iteration it: layer-0 WGs (0..63) run t = it; layer-1 WGs (64..127) run
// t = it-1. One device-wide barrier per iteration => T+1 = 513 barriers.
// ---------------------------------------------------------------------------
__global__ __launch_bounds__(256)
void lstm_persistent_kernel(const bf16_t* __restrict__ Xb,     // [T,32,1024] bf16
                            const bf16_t* __restrict__ Wihb,   // [L,4096,1024] bf16
                            const bf16_t* __restrict__ Whhb,   // [L,4096,1024] bf16
                            const float*  __restrict__ bih,    // [L,4096]
                            const float*  __restrict__ bhh,    // [L,4096]
                            float*        __restrict__ cstate, // [L,32,1024] f32
                            bf16_t*       __restrict__ hb,     // [2,L,32,1024] bf16
                            float*        __restrict__ out,    // [T,32,1024]++hT++cT
                            unsigned*     __restrict__ bar)    // {count, gen}
{
    __shared__ float gtile[8 * 256];

    const int wg    = blockIdx.x;
    const int layer = wg >> 6;           // 0 or 1
    const int n0    = (wg & 63) * 16;
    const size_t BH = (size_t)BATCH * HID;

    const bf16_t* Wih_l = Wihb + (size_t)layer * GATES * INP;
    const bf16_t* Whh_l = Whhb + (size_t)layer * GATES * HID;
    const float*  bih_l = bih + (size_t)layer * GATES;
    const float*  bhh_l = bhh + (size_t)layer * GATES;
    float*        cst_l = cstate + (size_t)layer * BH;

    for (int it = 0; it <= T_STEPS; ++it) {
        const int t = (layer == 0) ? it : it - 1;
        if (t >= 0 && t < T_STEPS) {
            const bf16_t* xin = (layer == 0)
                ? Xb + (size_t)t * BATCH * INP
                : hb + ((size_t)(t & 1) * NLAYER + 0) * BH;      // layer0 h at t
            const bf16_t* hprev = hb + ((size_t)((t + 1) & 1) * NLAYER + layer) * BH;
            bf16_t*       hbout = hb + ((size_t)(t & 1) * NLAYER + layer) * BH;
            float* memout = (layer == 1) ? out + (size_t)t * BH : nullptr;
            float* outT_h = (t == T_STEPS - 1)
                ? out + (size_t)T_STEPS * BH + (size_t)layer * BH : nullptr;
            float* outT_c = (t == T_STEPS - 1)
                ? out + (size_t)T_STEPS * BH + (size_t)NLAYER * BH + (size_t)layer * BH
                : nullptr;
            wg_cell(xin, hprev, Wih_l, Whh_l, bih_l, bhh_l, cst_l,
                    hbout, memout, outT_h, outT_c, gtile, n0);
        }
        grid_sync(&bar[0], &bar[1], (unsigned)it);
    }
}

__global__ void cvt_f32_bf16_kernel(const float* __restrict__ src,
                                    bf16_t* __restrict__ dst, int n) {
    int i = blockIdx.x * blockDim.x + threadIdx.x;
    int stride = gridDim.x * blockDim.x;
    for (; i < n; i += stride) dst[i] = (bf16_t)src[i];
}

__global__ void init_states_kernel(const float* __restrict__ h0,
                                   const float* __restrict__ c0,
                                   float* __restrict__ cst,
                                   bf16_t* __restrict__ hb_phase1,
                                   unsigned* __restrict__ bar, int n) {
    int i = blockIdx.x * blockDim.x + threadIdx.x;
    int stride = gridDim.x * blockDim.x;
    if (blockIdx.x == 0 && threadIdx.x == 0) { bar[0] = 0u; bar[1] = 0u; }
    for (; i < n; i += stride) {
        cst[i] = c0[i];
        hb_phase1[i] = (bf16_t)h0[i];
    }
}

extern "C" void kernel_launch(void* const* d_in, const int* in_sizes, int n_in,
                              void* d_out, int out_size, void* d_ws, size_t ws_size,
                              hipStream_t stream) {
    (void)in_sizes; (void)n_in; (void)out_size; (void)ws_size;

    const float* x   = (const float*)d_in[0]; // [T,B,IN]
    const float* h0  = (const float*)d_in[1]; // [L,B,H]
    const float* c0  = (const float*)d_in[2]; // [L,B,H]
    const float* Wih = (const float*)d_in[3]; // [L,4H,IN]
    const float* Whh = (const float*)d_in[4]; // [L,4H,H]
    const float* bih = (const float*)d_in[5]; // [L,4H]
    const float* bhh = (const float*)d_in[6]; // [L,4H]
    float* out = (float*)d_out;               // [T,B,H] ++ hT[L,B,H] ++ cT[L,B,H]

    char* ws = (char*)d_ws;
    size_t off = 0;
    auto wsalloc = [&](size_t bytes) -> void* {
        void* p = ws + off;
        off = (off + bytes + 255) & ~(size_t)255;
        return p;
    };

    const size_t BH  = (size_t)BATCH * HID;
    const size_t LBH = (size_t)NLAYER * BH;

    bf16_t*   Xb     = (bf16_t*)wsalloc((size_t)T_STEPS * BATCH * INP * 2); // 32 MB
    bf16_t*   Wihb   = (bf16_t*)wsalloc((size_t)NLAYER * GATES * INP * 2);  // 16 MB
    bf16_t*   Whhb   = (bf16_t*)wsalloc((size_t)NLAYER * GATES * HID * 2);  // 16 MB
    bf16_t*   hb     = (bf16_t*)wsalloc((size_t)2 * LBH * 2);               // 256 KB
    float*    cstate = (float*)wsalloc(LBH * 4);                            // 256 KB
    unsigned* bar    = (unsigned*)wsalloc(256);

    // --- precision conversion + state init pre-pass ---
    {
        const int nX = T_STEPS * BATCH * INP;
        cvt_f32_bf16_kernel<<<(nX + 255) / 256, 256, 0, stream>>>(x, Xb, nX);
        const int nW = NLAYER * GATES * INP;
        cvt_f32_bf16_kernel<<<(nW + 255) / 256, 256, 0, stream>>>(Wih, Wihb, nW);
        cvt_f32_bf16_kernel<<<(nW + 255) / 256, 256, 0, stream>>>(Whh, Whhb, nW);
        const int nS = (int)LBH;
        // t=0 reads h_prev from phase ((0-1)&1)==1 -> init phase-1 block
        init_states_kernel<<<(nS + 255) / 256, 256, 0, stream>>>(
            h0, c0, cstate, hb + LBH, bar, nS);
    }

    // --- one persistent kernel runs the full 512-step, 2-layer recurrence ---
    lstm_persistent_kernel<<<NWG, 256, 0, stream>>>(
        Xb, Wihb, Whhb, bih, bhh, cstate, hb, out, bar);
}
